// MistralAttention_73933567034006
// MI455X (gfx1250) — matrix operand order repaired and checked by reading, not compiled, not run on hardware
//
#include <hip/hip_runtime.h>

#define SEQ    2048
#define HIDDEN 4096
#define NH     32
#define NKV    8
#define HD     128
#define QW     (NH * HD)   // 4096
#define KVW    (NKV * HD)  // 1024

typedef __attribute__((ext_vector_type(16))) _Float16 v16h;
typedef __attribute__((ext_vector_type(8)))  float    v8f;

static __device__ __forceinline__ v8f wmma_f16(v16h a, v16h b, v8f c) {
  // 8 args: (neg_a, A, neg_b, B, c_mod, C, reuse_a, reuse_b) -> v_wmma_f32_16x16x32_f16
  return __builtin_amdgcn_wmma_f32_16x16x32_f16(false, a, false, b, (short)0, c, false, false);
}

// ---- per-lane fragment loaders -------------------------------------------
// A (16x32 f16): lanes L and L+16 hold row (L&15); K = koff + 0..15 contiguous,
// koff = 16*(lane>=16). Caller passes p = rowBase + kb + koff.
static __device__ __forceinline__ v16h ldA_f32(const float* __restrict__ p) {
  v16h r;
#pragma unroll
  for (int i = 0; i < 16; ++i) r[i] = (_Float16)p[i];
  return r;
}
static __device__ __forceinline__ v16h ldA_f16(const _Float16* __restrict__ p) {
  v16h r;
#pragma unroll
  for (int i = 0; i < 16; ++i) r[i] = p[i];
  return r;
}
// B (32x16): lane holds column (lane&15); K = 16*(lane>=16) + 0..15 contiguous.
// Caller passes p = colBase + kb + 16*(lane>=16).
static __device__ __forceinline__ v16h ldB_f32(const float* __restrict__ p) {
  v16h r;
#pragma unroll
  for (int i = 0; i < 16; ++i) r[i] = (_Float16)p[i];
  return r;
}
static __device__ __forceinline__ v16h ldB_f16(const _Float16* __restrict__ p) {
  v16h r;
#pragma unroll
  for (int i = 0; i < 16; ++i) r[i] = p[i];
  return r;
}

// ---- GEMM: Y = X @ W^T + bias, f32 inputs, f16 WMMA, f32 accumulate ------
// One wave per 32x64 output tile. grid = (N/64, M/32), block = 32.
// TRANS=true stores Y transposed (Y_t[n][m], leading dim M) with contiguous
// per-lane 8-row runs -> vectorized b128 stores.
template <typename OutT, bool TRANS>
__global__ void __launch_bounds__(32)
gemm_xwt_f16wmma(const float* __restrict__ X, const float* __restrict__ W,
                 const float* __restrict__ bias, OutT* __restrict__ Y,
                 int M, int N, int K) {
  const int lane = threadIdx.x;
  const int col  = lane & 15;
  const int half = lane >> 4;
  const int koff = half * 16;
  const int row0 = blockIdx.y * 32;
  const int col0 = blockIdx.x * 64;

  v8f c[2][4];
#pragma unroll
  for (int i = 0; i < 2; ++i)
#pragma unroll
    for (int j = 0; j < 4; ++j) c[i][j] = (v8f)0.0f;

  const float* arow[2];
  const float* brow[4];
#pragma unroll
  for (int i = 0; i < 2; ++i) arow[i] = X + (size_t)(row0 + i * 16 + col) * K;
#pragma unroll
  for (int j = 0; j < 4; ++j) brow[j] = W + (size_t)(col0 + j * 16 + col) * K;

  for (int kb = 0; kb < K; kb += 32) {
    v16h a[2], b[4];
#pragma unroll
    for (int i = 0; i < 2; ++i) {
      a[i] = ldA_f32(arow[i] + kb + koff);
      __builtin_prefetch(arow[i] + kb + 64, 0, 0);  // global_prefetch_b8
    }
#pragma unroll
    for (int j = 0; j < 4; ++j) {
      b[j] = ldB_f32(brow[j] + kb + half * 16);
      __builtin_prefetch(brow[j] + kb + 64, 0, 0);
    }
#pragma unroll
    for (int i = 0; i < 2; ++i)
#pragma unroll
      for (int j = 0; j < 4; ++j) c[i][j] = wmma_f16(a[i], b[j], c[i][j]);
  }

  // C/D layout: lane<16 -> rows r, N=lane ; lane>=16 -> rows r+8, N=lane-16
#pragma unroll
  for (int j = 0; j < 4; ++j) {
    const float bj = bias[col0 + j * 16 + col];
#pragma unroll
    for (int i = 0; i < 2; ++i) {
      if constexpr (TRANS) {
        // Y_t[n][m]: 8 consecutive rows per lane -> contiguous 16B run
        OutT* yp = Y + (size_t)(col0 + j * 16 + col) * M + row0 + i * 16 + half * 8;
#pragma unroll
        for (int r = 0; r < 8; ++r) yp[r] = (OutT)(c[i][j][r] + bj);
      } else {
#pragma unroll
        for (int r = 0; r < 8; ++r) {
          const int row = row0 + i * 16 + r + half * 8;
          Y[(size_t)row * N + col0 + j * 16 + col] = (OutT)(c[i][j][r] + bj);
        }
      }
    }
  }
}

// ---- fused causal flash attention (GQA 32q/8kv heads, HD=128) ------------
// One wave per (16-query block, head). grid = (SEQ/16, NH), block = 32.
// Kh is [SEQ][KVW] row-major; Vt is [KVW][SEQ] (transposed by its GEMM) so
// both K and V fragments are contiguous 2x b128 loads per lane.
__global__ void __launch_bounds__(32)
flash_attn_f16wmma(const _Float16* __restrict__ Qh, const _Float16* __restrict__ Kh,
                   const _Float16* __restrict__ Vt, float* __restrict__ H) {
  __shared__ _Float16 Pt[16 * 32];  // P tile staging, row-major [m][k]
  const int lane = threadIdx.x;
  const int col  = lane & 15;
  const int half = lane >> 4;
  const int koff = half * 16;
  const int q0   = blockIdx.x * 16;
  const int head = blockIdx.y;
  const int kvh  = head >> 2;
  const float sm = 0.088388347648318447f;  // 1/sqrt(128)

  // Q fragments (A layout), 4 k-steps over HD=128, kept resident
  v16h qf[4];
  {
    const _Float16* qb = Qh + (size_t)(q0 + col) * QW + head * HD;
#pragma unroll
    for (int t = 0; t < 4; ++t) qf[t] = ldA_f16(qb + t * 32 + koff);
  }

  v8f o[8];
#pragma unroll
  for (int t = 0; t < 8; ++t) o[t] = (v8f)0.0f;
  float mrun[8], lrun[8];
#pragma unroll
  for (int r = 0; r < 8; ++r) { mrun[r] = -3.0e38f; lrun[r] = 0.0f; }

  for (int kb0 = 0; kb0 <= q0 + 15; kb0 += 32) {
    // ---- scores S = Q K^T, two 16x16 key tiles ----
    // batch all 8 B-fragment loads into distinct regs before the WMMA chain
    v16h bk[2][4];
#pragma unroll
    for (int j = 0; j < 2; ++j) {
      const _Float16* kbp = Kh + (size_t)(kb0 + j * 16 + col) * KVW + kvh * HD;
#pragma unroll
      for (int t = 0; t < 4; ++t) bk[j][t] = ldB_f16(kbp + t * 32 + half * 16);
    }
    v8f s[2];
#pragma unroll
    for (int j = 0; j < 2; ++j) {
      v8f acc = (v8f)0.0f;
#pragma unroll
      for (int t = 0; t < 4; ++t) acc = wmma_f16(qf[t], bk[j][t], acc);
      s[j] = acc;
    }
    // ---- scale + causal mask (element r -> query row q0+r+8*half) ----
#pragma unroll
    for (int j = 0; j < 2; ++j) {
      const int key = kb0 + j * 16 + col;
#pragma unroll
      for (int r = 0; r < 8; ++r) {
        const int q = q0 + r + half * 8;
        const float v = s[j][r] * sm;
        s[j][r] = (q >= key) ? v : -3.0e38f;
      }
    }
    // ---- online softmax: row stats live across the 16 lanes of each half ----
    float mx[8];
#pragma unroll
    for (int r = 0; r < 8; ++r) mx[r] = fmaxf(s[0][r], s[1][r]);
#pragma unroll
    for (int d = 1; d < 16; d <<= 1)
#pragma unroll
      for (int r = 0; r < 8; ++r) mx[r] = fmaxf(mx[r], __shfl_xor(mx[r], d));
    float scale[8], mnew[8], sum[8];
#pragma unroll
    for (int r = 0; r < 8; ++r) {
      mnew[r]  = fmaxf(mrun[r], mx[r]);
      scale[r] = __expf(mrun[r] - mnew[r]);
      mrun[r]  = mnew[r];
      s[0][r]  = __expf(s[0][r] - mnew[r]);
      s[1][r]  = __expf(s[1][r] - mnew[r]);
      sum[r]   = s[0][r] + s[1][r];
    }
#pragma unroll
    for (int d = 1; d < 16; d <<= 1)
#pragma unroll
      for (int r = 0; r < 8; ++r) sum[r] += __shfl_xor(sum[r], d);
#pragma unroll
    for (int r = 0; r < 8; ++r) lrun[r] = lrun[r] * scale[r] + sum[r];
#pragma unroll
    for (int t = 0; t < 8; ++t)
#pragma unroll
      for (int r = 0; r < 8; ++r) o[t][r] *= scale[r];

    // ---- C-layout P -> LDS (row-major 16x32 f16) -> A-layout fragment ----
#pragma unroll
    for (int j = 0; j < 2; ++j)
#pragma unroll
      for (int r = 0; r < 8; ++r)
        Pt[(r + half * 8) * 32 + j * 16 + col] = (_Float16)s[j][r];
    asm volatile("s_wait_dscnt 0" ::: "memory");  // LDS in-order per wave; fence store->load
    const v16h pf = ldA_f16(&Pt[col * 32 + koff]);

    // ---- O += P * V, 8 d-tiles of 16; V^T makes each fragment 2x b128 ----
#pragma unroll
    for (int tb = 0; tb < 8; tb += 4) {
      v16h vf[4];
#pragma unroll
      for (int t = 0; t < 4; ++t)
        vf[t] = ldB_f16(Vt + (size_t)(kvh * HD + (tb + t) * 16 + col) * SEQ + kb0 + half * 16);
#pragma unroll
      for (int t = 0; t < 4; ++t) o[tb + t] = wmma_f16(pf, vf[t], o[tb + t]);
    }
    asm volatile("s_wait_dscnt 0" ::: "memory");  // keep Pt stable until reads retire
  }

  float inv[8];
#pragma unroll
  for (int r = 0; r < 8; ++r) inv[r] = 1.0f / lrun[r];
#pragma unroll
  for (int t = 0; t < 8; ++t)
#pragma unroll
    for (int r = 0; r < 8; ++r)
      H[(size_t)(q0 + r + half * 8) * QW + head * HD + t * 16 + col] = o[t][r] * inv[r];
}

extern "C" void kernel_launch(void* const* d_in, const int* in_sizes, int n_in,
                              void* d_out, int out_size, void* d_ws, size_t ws_size,
                              hipStream_t stream) {
  (void)in_sizes; (void)n_in; (void)out_size; (void)ws_size;
  const float* X  = (const float*)d_in[0];
  // d_in[1] = position_ids (int64) == arange(SEQ); causal mask uses implicit positions
  const float* Wq = (const float*)d_in[2];
  const float* bq = (const float*)d_in[3];
  const float* Wk = (const float*)d_in[4];
  const float* bk = (const float*)d_in[5];
  const float* Wv = (const float*)d_in[6];
  const float* bv = (const float*)d_in[7];
  const float* Wo = (const float*)d_in[8];
  const float* bo = (const float*)d_in[9];
  float* out = (float*)d_out;

  // workspace: Qh 16MB (f16) | Kh 4MB | Vt 4MB (transposed) | H 32MB (f32) -> 56MB
  _Float16* Qh = (_Float16*)d_ws;
  _Float16* Kh = Qh + (size_t)SEQ * QW;
  _Float16* Vt = Kh + (size_t)SEQ * KVW;
  float*    Hb = (float*)(Vt + (size_t)SEQ * KVW);

  dim3 blk(32, 1, 1);
  gemm_xwt_f16wmma<_Float16, false><<<dim3(QW / 64,  SEQ / 32), blk, 0, stream>>>(X, Wq, bq, Qh, SEQ, QW,  HIDDEN);
  gemm_xwt_f16wmma<_Float16, false><<<dim3(KVW / 64, SEQ / 32), blk, 0, stream>>>(X, Wk, bk, Kh, SEQ, KVW, HIDDEN);
  gemm_xwt_f16wmma<_Float16, true ><<<dim3(KVW / 64, SEQ / 32), blk, 0, stream>>>(X, Wv, bv, Vt, SEQ, KVW, HIDDEN);
  flash_attn_f16wmma<<<dim3(SEQ / 16, NH), blk, 0, stream>>>(Qh, Kh, Vt, Hb);
  gemm_xwt_f16wmma<float, false><<<dim3(HIDDEN / 64, SEQ / 32), blk, 0, stream>>>(Hb, Wo, bo, out, SEQ, HIDDEN, HIDDEN);
}
